// SchemaAugmentedSOMT_31387620999553
// MI455X (gfx1250) — compile-verified, matrix operand order checked by reading, not run on hardware
//
#include <hip/hip_runtime.h>
#include <hip/hip_bf16.h>

// ---------------------------------------------------------------------------
// SchemaAugmentedSOMT forward for MI455X (gfx1250, wave32, WMMA).
// Core math (projections, FFN, vocab logits) runs on v_wmma_f32_16x16x32_f16
// with fp32 accumulation; A and B tiles are staged into double-buffered LDS
// with the CDNA5 async global->LDS path (ASYNCcnt), overlapping DMA with the
// WMMA math; long-tail ops (<3% FLOPs) are VALU kernels.
// ---------------------------------------------------------------------------

typedef _Float16 f16;
typedef __attribute__((ext_vector_type(16))) _Float16 v16h;
typedef __attribute__((ext_vector_type(8)))  float    v8f;

#define BB     2
#define LL     1024
#define DD     512
#define HH     8
#define DH     64
#define NLAYER 4
#define VV     50257
#define MEMN   512
#define SS     64
#define BL     (BB*LL)
#define NCOMB  (MEMN+LL)   // 1536
#define NREAD  (MEMN+SS)   // 576

#define LDSPAD 40          // f16 per LDS row (32 data + 8 pad) -> 80B stride

// ===========================================================================
// WMMA GEMM: C[M,N] = act( A[M,K] @ B[K,N] + bias ), B given as Bt[N,K]
// (row-major "B-transposed" == N-major), A row-major, both f16, fp32 accum.
// Block: 256 threads (8 waves) computing a 64(M) x 128(N) tile; wave (wm,wn)
// owns a 32x32 patch = 2x2 WMMA 16x16 tiles (A/B fragment reuse).
// K must be a multiple of 32. act: 0=none, 1=relu, 2=sigmoid.
// Both tiles staged via per-lane GLOBAL_LOAD_ASYNC_TO_LDS_B128 into
// double-buffered LDS; out-of-range A/B rows are clamped (their products only
// reach C rows/cols that the guarded epilogue never stores).
// ===========================================================================
__global__ __launch_bounds__(256)
void k_gemm(const f16* __restrict__ A, const f16* __restrict__ Bt,
            const float* __restrict__ bias,
            float* __restrict__ Cf, f16* __restrict__ Ch,
            int M, int N, int K, int act)
{
  __shared__ f16 As[2][64][LDSPAD];
  __shared__ f16 Bs[2][128][LDSPAD];

  const int m0   = blockIdx.y * 64;
  const int n0   = blockIdx.x * 128;
  const int tid  = (int)threadIdx.x;
  const int wave = tid >> 5;
  const int wm   = wave >> 2;          // 0..1  (M sub-block)
  const int wn   = wave & 3;           // 0..3  (N sub-block)
  const int lane = tid & 31;
  const int hi   = lane >> 4;          // half-wave select (ISA 16-bit layout)
  const int row  = lane & 15;

  // per-thread staging coordinates (addresses clamped into range)
  const int ar = tid >> 2;             // 0..63  A row
  const int ac = (tid & 3) * 8;        // 0,8,16,24 (8 f16 = 16B)
  int agm = m0 + ar; if (agm >= M) agm = M - 1;
  const int br = tid >> 1;             // 0..127 B row (column of B)
  const int bc = (tid & 1) * 16;       // 0 or 16 (16 f16 = 2x16B)
  int bgn = n0 + br; if (bgn >= N) bgn = N - 1;

  const f16* aptr = A  + (size_t)agm * K + ac;
  const f16* bptr = Bt + (size_t)bgn * K + bc;
  unsigned a_lds[2], b_lds[2];
  a_lds[0] = (unsigned)(unsigned long long)(&As[0][ar][ac]);
  a_lds[1] = (unsigned)(unsigned long long)(&As[1][ar][ac]);
  b_lds[0] = (unsigned)(unsigned long long)(&Bs[0][br][bc]);
  b_lds[1] = (unsigned)(unsigned long long)(&Bs[1][br][bc]);

  v8f acc[2][2] = {};

  // warm L2 for slab 1 while the prologue DMA runs
  if (K > 32) {
    __builtin_prefetch(aptr + 32, 0, 1);
    __builtin_prefetch(bptr + 32, 0, 1);
  }
  // prologue: stage K-slab 0 into buffer 0
  {
    unsigned long long ga = (unsigned long long)aptr;
    unsigned long long gb = (unsigned long long)bptr;
    asm volatile("global_load_async_to_lds_b128 %0, %1, off"
                 :: "v"(a_lds[0]), "v"(ga) : "memory");
    asm volatile("global_load_async_to_lds_b128 %0, %1, off"
                 :: "v"(b_lds[0]), "v"(gb) : "memory");
    asm volatile("global_load_async_to_lds_b128 %0, %1, off"
                 :: "v"(b_lds[0] + 16u), "v"(gb + 16ull) : "memory");
  }
  asm volatile("s_wait_asynccnt 0x0" ::: "memory");
  __syncthreads();

  for (int k0 = 0; k0 < K; k0 += 32) {
    const int cur = (k0 >> 5) & 1;
    // stage next K-slab into the other buffer (overlaps with WMMA below)
    if (k0 + 32 < K) {
      const int nxt = cur ^ 1;
      unsigned long long ga = (unsigned long long)(aptr + (k0 + 32));
      unsigned long long gb = (unsigned long long)(bptr + (k0 + 32));
      asm volatile("global_load_async_to_lds_b128 %0, %1, off"
                   :: "v"(a_lds[nxt]), "v"(ga) : "memory");
      asm volatile("global_load_async_to_lds_b128 %0, %1, off"
                   :: "v"(b_lds[nxt]), "v"(gb) : "memory");
      asm volatile("global_load_async_to_lds_b128 %0, %1, off"
                   :: "v"(b_lds[nxt] + 16u), "v"(gb + 16ull) : "memory");
    }

    // ---- build fragments per ISA 16-bit layout (K-pairs per VGPR) ----
    union Frag { v16h v; unsigned int u[8]; } af[2], bf[2];
#pragma unroll
    for (int sub = 0; sub < 2; ++sub) {
      const int arow = wm * 32 + sub * 16 + row;
      const int brow = wn * 32 + sub * 16 + row;
#pragma unroll
      for (int vr = 0; vr < 8; ++vr) {
        int kk = ((vr < 4) ? (vr * 2) : (16 + (vr - 4) * 2)) + hi * 8;
        af[sub].u[vr] = *(const unsigned int*)&As[cur][arow][kk];
        bf[sub].u[vr] = *(const unsigned int*)&Bs[cur][brow][kk];
      }
    }
#pragma unroll
    for (int mi = 0; mi < 2; ++mi)
#pragma unroll
      for (int ni = 0; ni < 2; ++ni)
        acc[mi][ni] = __builtin_amdgcn_wmma_f32_16x16x32_f16(
            false, af[mi].v, false, bf[ni].v, (short)0, acc[mi][ni],
            false, false);

    // next tile's DMA must be LDS-visible to all waves before the next pass
    asm volatile("s_wait_asynccnt 0x0" ::: "memory");
    __syncthreads();
  }

  // ---- epilogue: D layout: col n = lane&15, rows m = r + 8*hi ----
#pragma unroll
  for (int mi = 0; mi < 2; ++mi) {
#pragma unroll
    for (int ni = 0; ni < 2; ++ni) {
      const int n = n0 + wn * 32 + ni * 16 + row;
      const int mbase = m0 + wm * 32 + mi * 16 + hi * 8;
      const float bv = (bias != nullptr && n < N) ? bias[n] : 0.0f;
#pragma unroll
      for (int r = 0; r < 8; ++r) {
        int m = mbase + r;
        if (m < M && n < N) {
          float v = acc[mi][ni][r] + bv;
          if (act == 1)      v = fmaxf(v, 0.0f);
          else if (act == 2) v = 1.0f / (1.0f + __expf(-v));
          if (Cf) Cf[(size_t)m * N + n] = v;
          if (Ch) Ch[(size_t)m * N + n] = (f16)v;
        }
      }
    }
  }
}

// ===========================================================================
// Conversions
// ===========================================================================
__global__ void k_f2h(const float* __restrict__ s, f16* __restrict__ d, size_t n)
{
  for (size_t i = blockIdx.x * (size_t)blockDim.x + threadIdx.x; i < n;
       i += (size_t)gridDim.x * blockDim.x)
    d[i] = (f16)s[i];
}

// src: nb batches of (K x N) row-major -> dst: nb batches of (N x K) f16
__global__ void k_f2h_T(const float* __restrict__ s, f16* __restrict__ d,
                        int K, int N, int nb)
{
  size_t per = (size_t)K * N, total = per * nb;
  for (size_t i = blockIdx.x * (size_t)blockDim.x + threadIdx.x; i < total;
       i += (size_t)gridDim.x * blockDim.x) {
    size_t b = i / per, r = i % per;
    int k = (int)(r / N), n = (int)(r % N);
    d[b * per + (size_t)n * K + k] = (f16)s[i];
  }
}

// ===========================================================================
// Embedding: x_emb = embed[x] + pos_embed[0,:L]; also seeds h.
// ===========================================================================
__global__ void k_embed(const int* __restrict__ x, const float* __restrict__ emb,
                        const float* __restrict__ pos,
                        float* xe_f, f16* xe_h, float* h_f, f16* h_h)
{
  size_t i = blockIdx.x * (size_t)blockDim.x + threadIdx.x;
  if (i >= (size_t)BL * DD) return;
  int d = (int)(i % DD);
  size_t tok = i / DD;
  int l = (int)(tok % LL);
  int t = x[tok];
  float v = emb[(size_t)t * DD + d] + pos[(size_t)l * DD + d];
  xe_f[i] = v; xe_h[i] = (f16)v; h_f[i] = v; h_h[i] = (f16)v;
}

// ===========================================================================
// Causal attention, one block per (b,h,q). Dh=64. VALU (4.3 GFLOP total).
// ===========================================================================
__global__ __launch_bounds__(128)
void k_attn(const float* __restrict__ Q, const float* __restrict__ Kc,
            const float* __restrict__ Vc, float* __restrict__ Of,
            f16* __restrict__ Oh)
{
  __shared__ float qv[DH];
  __shared__ float sc[LL];
  __shared__ float rb[128];
  int bid = blockIdx.x;
  int q = bid % LL, h = (bid / LL) % HH, b = bid / (LL * HH);
  size_t base = (size_t)b * LL * DD + (size_t)h * DH;
  int tid = (int)threadIdx.x;
  if (tid < DH) qv[tid] = Q[base + (size_t)q * DD + tid];
  __syncthreads();
  const float scale = 0.125f;  // 1/sqrt(64)
  for (int j = tid; j < LL; j += 128) {
    if (j <= q) {
      const float* kp = Kc + base + (size_t)j * DD;
      float s = 0.0f;
      for (int d = 0; d < DH; ++d) s += qv[d] * kp[d];
      sc[j] = s * scale;
    } else sc[j] = -1e30f;
  }
  __syncthreads();
  float lm = -1e30f;
  for (int j = tid; j < LL; j += 128) lm = fmaxf(lm, sc[j]);
  rb[tid] = lm; __syncthreads();
  for (int o = 64; o; o >>= 1) { if (tid < o) rb[tid] = fmaxf(rb[tid], rb[tid + o]); __syncthreads(); }
  float m = rb[0]; __syncthreads();
  float ls = 0.0f;
  for (int j = tid; j < LL; j += 128) { float e = __expf(sc[j] - m); sc[j] = e; ls += e; }
  rb[tid] = ls; __syncthreads();
  for (int o = 64; o; o >>= 1) { if (tid < o) rb[tid] += rb[tid + o]; __syncthreads(); }
  float inv = 1.0f / rb[0];
  __syncthreads();
  if (tid < DH) {
    const float* vp = Vc + base + tid;
    float o = 0.0f;
    for (int j = 0; j <= q; ++j) o += sc[j] * vp[(size_t)j * DD];
    o *= inv;
    Of[base + (size_t)q * DD + tid] = o;
    Oh[base + (size_t)q * DD + tid] = (f16)o;
  }
}

// ===========================================================================
// out = LN(x + y) * w + b ; one block (256 thr) per row of DD=512.
// ===========================================================================
__global__ __launch_bounds__(256)
void k_add_ln(const float* __restrict__ x, const float* __restrict__ y,
              const float* __restrict__ w, const float* __restrict__ bsh,
              float* __restrict__ of, f16* __restrict__ oh)
{
  __shared__ float rb[256];
  int row = blockIdx.x, tid = (int)threadIdx.x;
  const float* xr = x + (size_t)row * DD;
  const float* yr = y + (size_t)row * DD;
  float v0 = xr[tid] + yr[tid], v1 = xr[tid + 256] + yr[tid + 256];
  rb[tid] = v0 + v1; __syncthreads();
  for (int o = 128; o; o >>= 1) { if (tid < o) rb[tid] += rb[tid + o]; __syncthreads(); }
  float mean = rb[0] * (1.0f / DD); __syncthreads();
  float c0 = v0 - mean, c1 = v1 - mean;
  rb[tid] = c0 * c0 + c1 * c1; __syncthreads();
  for (int o = 128; o; o >>= 1) { if (tid < o) rb[tid] += rb[tid + o]; __syncthreads(); }
  float inv = rsqrtf(rb[0] * (1.0f / DD) + 1e-5f);
  float r0 = c0 * inv * w[tid] + bsh[tid];
  float r1 = c1 * inv * w[tid + 256] + bsh[tid + 256];
  size_t o0 = (size_t)row * DD + tid;
  of[o0] = r0; of[o0 + 256] = r1;
  oh[o0] = (f16)r0; oh[o0 + 256] = (f16)r1;
}

// ===========================================================================
// Softmax entropy over V per row: ent = m + log(sum e^{x-m}) - (sum x e)/sum e
// ===========================================================================
__global__ __launch_bounds__(256)
void k_entropy(const float* __restrict__ logits, float* __restrict__ ent)
{
  __shared__ float rb[256];
  int row = blockIdx.x, tid = (int)threadIdx.x;
  const float* lp = logits + (size_t)row * VV;
  float lm = -3e38f;
  for (int j = tid; j < VV; j += 256) lm = fmaxf(lm, lp[j]);
  rb[tid] = lm; __syncthreads();
  for (int o = 128; o; o >>= 1) { if (tid < o) rb[tid] = fmaxf(rb[tid], rb[tid + o]); __syncthreads(); }
  float m = rb[0]; __syncthreads();
  float s = 0.0f, u = 0.0f;
  for (int j = tid; j < VV; j += 256) {
    float x = lp[j], e = __expf(x - m);
    s += e; u += x * e;
  }
  rb[tid] = s; __syncthreads();
  for (int o = 128; o; o >>= 1) { if (tid < o) rb[tid] += rb[tid + o]; __syncthreads(); }
  s = rb[0]; __syncthreads();
  rb[tid] = u; __syncthreads();
  for (int o = 128; o; o >>= 1) { if (tid < o) rb[tid] += rb[tid + o]; __syncthreads(); }
  u = rb[0];
  if (tid == 0) ent[row] = m + logf(s) - u / s;
}

// ent_n + per-batch stats: stats[2b]=mean(ent_n), stats[2b+1]=max(ent_n)
__global__ __launch_bounds__(256)
void k_entn(const float* __restrict__ ent, float* __restrict__ ent_n,
            float* __restrict__ stats)
{
  __shared__ float rb[256];
  int b = blockIdx.x, tid = (int)threadIdx.x;
  const float* e = ent + (size_t)b * LL;
  float lmin = 3e38f, lmax = -3e38f, lsum = 0.0f;
  for (int l = tid; l < LL; l += 256) { float v = e[l]; lmin = fminf(lmin, v); lmax = fmaxf(lmax, v); lsum += v; }
  rb[tid] = lmin; __syncthreads();
  for (int o = 128; o; o >>= 1) { if (tid < o) rb[tid] = fminf(rb[tid], rb[tid + o]); __syncthreads(); }
  float emin = rb[0]; __syncthreads();
  rb[tid] = lmax; __syncthreads();
  for (int o = 128; o; o >>= 1) { if (tid < o) rb[tid] = fmaxf(rb[tid], rb[tid + o]); __syncthreads(); }
  float emax = rb[0]; __syncthreads();
  rb[tid] = lsum; __syncthreads();
  for (int o = 128; o; o >>= 1) { if (tid < o) rb[tid] += rb[tid + o]; __syncthreads(); }
  float tot = rb[0];
  float inv = 1.0f / (emax - emin + 1e-12f);
  for (int l = tid; l < LL; l += 256) ent_n[(size_t)b * LL + l] = (e[l] - emin) * inv;
  if (tid == 0) { stats[b * 2 + 0] = (tot / LL - emin) * inv; stats[b * 2 + 1] = (emax - emin) * inv; }
}

// ===========================================================================
// queries = (enc@qW + qb) * (1 + unc)
// ===========================================================================
__global__ void k_queries(const float* __restrict__ proj, const float* __restrict__ unc,
                          float* __restrict__ qout)
{
  size_t i = blockIdx.x * (size_t)blockDim.x + threadIdx.x;
  if (i >= (size_t)BL * DD) return;
  qout[i] = proj[i] * (1.0f + unc[i]);
}

// ===========================================================================
// Threshold MLP: sigmoid( relu([enc,avg,mx] @ W1(514x64) + b1) @ W2 + b2 )
// ===========================================================================
__global__ __launch_bounds__(64)
void k_thr(const float* __restrict__ enc, const float* __restrict__ stats,
           const float* __restrict__ W1, const float* __restrict__ b1,
           const float* __restrict__ W2, const float* __restrict__ b2,
           float* __restrict__ out)
{
  __shared__ float rb[64];
  int row = blockIdx.x, j = (int)threadIdx.x, b = row / LL;
  const float* xr = enc + (size_t)row * DD;
  float s = b1[j];
  for (int k = 0; k < DD; ++k) s += xr[k] * W1[k * 64 + j];
  s += stats[b * 2 + 0] * W1[DD * 64 + j] + stats[b * 2 + 1] * W1[(DD + 1) * 64 + j];
  s = fmaxf(s, 0.0f);
  rb[j] = s * W2[j]; __syncthreads();
  for (int o = 32; o; o >>= 1) { if (j < o) rb[j] += rb[j + o]; __syncthreads(); }
  if (j == 0) out[row] = 1.0f / (1.0f + __expf(-(rb[0] + b2[0])));
}

// Generic 512->64->1 MLP (relu hidden), one block (64 thr) per row.
__global__ __launch_bounds__(64)
void k_mlp512_64_1(const float* __restrict__ X,
                   const float* __restrict__ W1, const float* __restrict__ b1,
                   const float* __restrict__ W2, const float* __restrict__ b2,
                   float* __restrict__ out, int inner, int ostride, int ooff)
{
  __shared__ float rb[64];
  int row = blockIdx.x, j = (int)threadIdx.x;
  const float* xr = X + (size_t)row * DD;
  float s = b1[j];
  for (int k = 0; k < DD; ++k) s += xr[k] * W1[k * 64 + j];
  s = fmaxf(s, 0.0f);
  rb[j] = s * W2[j]; __syncthreads();
  for (int o = 32; o; o >>= 1) { if (j < o) rb[j] += rb[j + o]; __syncthreads(); }
  if (j == 0)
    out[(size_t)(row / inner) * ostride + ooff + (row % inner)] = rb[0] + b2[0];
}

// Generic 512->32->1 MLP; input row remapped via (in_inner,in_stride,in_off).
__global__ __launch_bounds__(32)
void k_mlp512_32_1(const float* __restrict__ X,
                   const float* __restrict__ W1, const float* __restrict__ b1,
                   const float* __restrict__ W2, const float* __restrict__ b2,
                   float* __restrict__ out, int in_inner, int in_stride,
                   int in_off, int sig)
{
  __shared__ float rb[32];
  int row = blockIdx.x, j = (int)threadIdx.x;
  size_t xrow = (size_t)(row / in_inner) * in_stride + in_off + (row % in_inner);
  const float* xr = X + xrow * DD;
  float s = b1[j];
  for (int k = 0; k < DD; ++k) s += xr[k] * W1[k * 32 + j];
  s = fmaxf(s, 0.0f);
  rb[j] = s * W2[j]; __syncthreads();
  for (int o = 16; o; o >>= 1) { if (j < o) rb[j] += rb[j + o]; __syncthreads(); }
  if (j == 0) {
    float v = rb[0] + b2[0];
    out[row] = sig ? (1.0f / (1.0f + __expf(-v))) : v;
  }
}

// mean over L: out[b,d] = mean_l enc[b,l,d]
__global__ void k_meanL(const float* __restrict__ enc, float* __restrict__ out)
{
  int i = blockIdx.x * blockDim.x + threadIdx.x;
  if (i >= BB * DD) return;
  int b = i / DD, d = i % DD;
  float s = 0.0f;
  for (int l = 0; l < LL; ++l) s += enc[((size_t)b * LL + l) * DD + d];
  out[i] = s * (1.0f / LL);
}

// importance adjust + mask
__global__ void k_impm(const float* __restrict__ imp_raw, const float* __restrict__ obs,
                       const float* __restrict__ age, const float* __restrict__ ent_n,
                       const float* __restrict__ thr, float* __restrict__ impA,
                       float* __restrict__ imp_m, float* __restrict__ validf)
{
  int i = blockIdx.x * blockDim.x + threadIdx.x;
  if (i >= BB * NCOMB) return;
  int b = i / NCOMB, n = i % NCOMB;
  float v = imp_raw[i];
  float valid = 1.0f;
  if (n < MEMN) {
    v -= (age[b * MEMN + n] + 1.0f) * obs[b * MEMN + n];
  } else {
    int l = n - MEMN;
    v += 0.01f * (1.0f - 0.9f * ((float)l / (float)(LL - 1)));
    valid = (ent_n[(size_t)b * LL + l] > thr[(size_t)b * LL + l]) ? 1.0f : 0.0f;
  }
  impA[i] = v;
  validf[i] = valid;
  imp_m[i] = (valid > 0.0f) ? v : -1e30f;
}

// per-batch softmax entropy of imp_m + l2/count stats
__global__ __launch_bounds__(256)
void k_regstats(const float* __restrict__ imp_m, const float* __restrict__ impA,
                const float* __restrict__ validf, float* __restrict__ stats2)
{
  __shared__ float rb[256];
  int b = blockIdx.x, tid = (int)threadIdx.x;
  const float* xm = imp_m + (size_t)b * NCOMB;
  float lm = -3e38f;
  for (int n = tid; n < NCOMB; n += 256) lm = fmaxf(lm, xm[n]);
  rb[tid] = lm; __syncthreads();
  for (int o = 128; o; o >>= 1) { if (tid < o) rb[tid] = fmaxf(rb[tid], rb[tid + o]); __syncthreads(); }
  float m = rb[0]; __syncthreads();
  float s = 0, u = 0, l2 = 0, cnt = 0;
  for (int n = tid; n < NCOMB; n += 256) {
    float x = xm[n], e = __expf(x - m);
    s += e; u += x * e;
    float va = validf[(size_t)b * NCOMB + n];
    float ia = impA[(size_t)b * NCOMB + n];
    l2 += va * ia * ia; cnt += va;
  }
  rb[tid] = s; __syncthreads();
  for (int o = 128; o; o >>= 1) { if (tid < o) rb[tid] += rb[tid + o]; __syncthreads(); }
  s = rb[0]; __syncthreads();
  rb[tid] = u; __syncthreads();
  for (int o = 128; o; o >>= 1) { if (tid < o) rb[tid] += rb[tid + o]; __syncthreads(); }
  u = rb[0]; __syncthreads();
  rb[tid] = l2; __syncthreads();
  for (int o = 128; o; o >>= 1) { if (tid < o) rb[tid] += rb[tid + o]; __syncthreads(); }
  l2 = rb[0]; __syncthreads();
  rb[tid] = cnt; __syncthreads();
  for (int o = 128; o; o >>= 1) { if (tid < o) rb[tid] += rb[tid + o]; __syncthreads(); }
  cnt = rb[0];
  if (tid == 0) {
    stats2[b * 4 + 0] = m + logf(s) - u / s;  // -sum p log p for this batch
    stats2[b * 4 + 1] = l2;
    stats2[b * 4 + 2] = cnt;
  }
}

// top-512 of 1536, descending (iterative argmax; prefers lower index on ties)
__global__ __launch_bounds__(256)
void k_topk(const float* __restrict__ imp_m, int* __restrict__ top_idx)
{
  __shared__ float vals[NCOMB];
  __shared__ float bv[256];
  __shared__ int   bi[256];
  int b = blockIdx.x, tid = (int)threadIdx.x;
  for (int n = tid; n < NCOMB; n += 256) vals[n] = imp_m[(size_t)b * NCOMB + n];
  __syncthreads();
  for (int r = 0; r < MEMN; ++r) {
    float best = -3e38f; int bidx = NCOMB;
    for (int n = tid; n < NCOMB; n += 256) {
      float v = vals[n];
      if (v > best || (v == best && n < bidx)) { best = v; bidx = n; }
    }
    bv[tid] = best; bi[tid] = bidx; __syncthreads();
    for (int o = 128; o; o >>= 1) {
      if (tid < o) {
        if (bv[tid + o] > bv[tid] || (bv[tid + o] == bv[tid] && bi[tid + o] < bi[tid])) {
          bv[tid] = bv[tid + o]; bi[tid] = bi[tid + o];
        }
      }
      __syncthreads();
    }
    if (tid == 0) { top_idx[b * MEMN + r] = bi[0]; vals[bi[0]] = -3e38f; }
    __syncthreads();
  }
}

// gather upd_k/upd_v/upd_age into d_out and read_k/read_v buffers
__global__ void k_gather(const int* __restrict__ top_idx,
                         const float* __restrict__ mem_k, const float* __restrict__ mem_v,
                         const float* __restrict__ age,
                         const float* __restrict__ cand_k, const float* __restrict__ cand_v,
                         float* __restrict__ updk_o, float* __restrict__ updv_o,
                         float* __restrict__ age_o,
                         float* __restrict__ readk_f, float* __restrict__ readv_f)
{
  size_t i = blockIdx.x * (size_t)blockDim.x + threadIdx.x;
  if (i >= (size_t)BB * MEMN * DD) return;
  int d = (int)(i % DD);
  int m = (int)((i / DD) % MEMN);
  int b = (int)(i / ((size_t)DD * MEMN));
  int idx = top_idx[b * MEMN + m];
  float kv, vv;
  if (idx < MEMN) {
    kv = mem_k[((size_t)b * MEMN + idx) * DD + d];
    vv = mem_v[((size_t)b * MEMN + idx) * DD + d];
  } else {
    int l = idx - MEMN;
    kv = cand_k[((size_t)b * LL + l) * DD + d];
    vv = cand_v[((size_t)b * LL + l) * DD + d];
  }
  updk_o[i] = kv; updv_o[i] = vv;
  size_t rrow = (size_t)b * NREAD + m;
  readk_f[rrow * DD + d] = kv;
  readv_f[rrow * DD + d] = vv;
  if (d == 0) age_o[b * MEMN + m] = (idx < MEMN) ? age[b * MEMN + idx] + 1.0f : 0.0f;
}

// routing: softmax(relu(upd_k@W1+b1)@W2+b2) over S=64; block 64 thr per row
__global__ __launch_bounds__(64)
void k_routing(const float* __restrict__ updk,
               const float* __restrict__ W1, const float* __restrict__ b1,
               const float* __restrict__ W2, const float* __restrict__ b2,
               float* __restrict__ routing)
{
  __shared__ float hid[64];
  __shared__ float rb[64];
  int row = blockIdx.x, j = (int)threadIdx.x;
  const float* xr = updk + (size_t)row * DD;
  float s = b1[j];
  for (int k = 0; k < DD; ++k) s += xr[k] * W1[k * 64 + j];
  hid[j] = fmaxf(s, 0.0f);
  __syncthreads();
  float o = b2[j];
  for (int t = 0; t < 64; ++t) o += hid[t] * W2[t * SS + j];
  rb[j] = o; __syncthreads();
  for (int t = 32; t; t >>= 1) { if (j < t) rb[j] = fmaxf(rb[j], rb[j + t]); __syncthreads(); }
  float m = rb[0]; __syncthreads();
  float e = __expf(o - m);
  rb[j] = e; __syncthreads();
  for (int t = 32; t; t >>= 1) { if (j < t) rb[j] += rb[j + t]; __syncthreads(); }
  routing[(size_t)row * SS + j] = e / rb[0];
}

// schema_reps[b,s,:] = sum_m routing[b,m,s]*upd_k[b,m,:]; fills read_k/read_v tails
__global__ void k_schema(const float* __restrict__ routing, const float* __restrict__ updk,
                         const float* __restrict__ schema_vals,
                         float* __restrict__ readk_f, float* __restrict__ readv_f)
{
  size_t i = blockIdx.x * (size_t)blockDim.x + threadIdx.x;
  if (i >= (size_t)BB * SS * DD) return;
  int d = (int)(i % DD);
  int s = (int)((i / DD) % SS);
  int b = (int)(i / ((size_t)DD * SS));
  const float* rp = routing + (size_t)b * MEMN * SS + s;
  const float* kp = updk + (size_t)b * MEMN * DD + d;
  float acc = 0.0f;
  for (int m = 0; m < MEMN; ++m) acc += rp[(size_t)m * SS] * kp[(size_t)m * DD];
  size_t rrow = (size_t)b * NREAD + MEMN + s;
  readk_f[rrow * DD + d] = acc;
  readv_f[rrow * DD + d] = schema_vals[(size_t)s * DD + d];
}

// retrieval attention over NREAD=576 slots per token; block 128 thr
__global__ __launch_bounds__(128)
void k_retrieve(const float* __restrict__ queries, const float* __restrict__ readk_f,
                const float* __restrict__ readv_f,
                float* __restrict__ ret_f, f16* __restrict__ ret_h)
{
  __shared__ float sc[NREAD];
  __shared__ float rb[128];
  int row = blockIdx.x, tid = (int)threadIdx.x;
  int b = row / LL;
  const float* qr = queries + (size_t)row * DD;
  const float inv_sqrt_d = 0.044194173824159216f;  // 1/sqrt(512)
  for (int n = tid; n < NREAD; n += 128) {
    const float* kp = readk_f + ((size_t)b * NREAD + n) * DD;
    float s = 0.0f;
    for (int d = 0; d < DD; ++d) s += qr[d] * kp[d];
    sc[n] = s * inv_sqrt_d;
  }
  __syncthreads();
  float lm = -3e38f;
  for (int n = tid; n < NREAD; n += 128) lm = fmaxf(lm, sc[n]);
  rb[tid] = lm; __syncthreads();
  for (int o = 64; o; o >>= 1) { if (tid < o) rb[tid] = fmaxf(rb[tid], rb[tid + o]); __syncthreads(); }
  float m = rb[0]; __syncthreads();
  float ls = 0.0f;
  for (int n = tid; n < NREAD; n += 128) { float e = __expf(sc[n] - m); sc[n] = e; ls += e; }
  rb[tid] = ls; __syncthreads();
  for (int o = 64; o; o >>= 1) { if (tid < o) rb[tid] += rb[tid + o]; __syncthreads(); }
  float inv = 1.0f / rb[0];
  __syncthreads();
  for (int d = tid; d < DD; d += 128) {
    const float* vp = readv_f + (size_t)b * NREAD * DD + d;
    float o = 0.0f;
    for (int n = 0; n < NREAD; ++n) o += sc[n] * vp[(size_t)n * DD];
    o *= inv;
    ret_f[(size_t)row * DD + d] = o;
    ret_h[(size_t)row * DD + d] = (f16)o;
  }
}

// fused = enc + (retrieved@fuse_W + fuse_b)   (second operand precomputed)
__global__ void k_fused(const float* __restrict__ enc, const float* __restrict__ proj,
                        f16* __restrict__ fused_h)
{
  size_t i = blockIdx.x * (size_t)blockDim.x + threadIdx.x;
  if (i >= (size_t)BL * DD) return;
  fused_h[i] = (f16)(enc[i] + proj[i]);
}

// scalars: entropy_reg, l2_imp, util.mean()
__global__ void k_finalize(const float* __restrict__ stats2,
                           const float* __restrict__ util_buf,
                           float* __restrict__ outp)
{
  if (threadIdx.x != 0 || blockIdx.x != 0) return;
  outp[0] = stats2[0] + stats2[4];
  outp[1] = (stats2[1] + stats2[5]) / (stats2[2] + stats2[6]);
  float us = 0.0f;
  for (int i = 0; i < BB * SS; ++i) us += util_buf[i];
  outp[2] = us / (float)(BB * SS);
}

// ===========================================================================
// Host orchestration
// ===========================================================================
static inline dim3 gemm_grid(int M, int N) { return dim3((N + 127) / 128, (M + 63) / 64); }

extern "C" void kernel_launch(void* const* d_in, const int* in_sizes, int n_in,
                              void* d_out, int out_size, void* d_ws, size_t ws_size,
                              hipStream_t stream)
{
  (void)in_sizes; (void)n_in; (void)out_size; (void)ws_size;

  // -------- inputs (setup_inputs dict order) --------
  const int*   x       = (const int*)  d_in[0];
  const float* mem_k   = (const float*)d_in[1];
  const float* mem_v   = (const float*)d_in[2];
  const float* mem_age = (const float*)d_in[3];
  const float* embed   = (const float*)d_in[4];
  const float* pos     = (const float*)d_in[5];
  const float* Wq = (const float*)d_in[6],  *Wk = (const float*)d_in[7];
  const float* Wv = (const float*)d_in[8],  *Wo = (const float*)d_in[9];
  const float* Wff1 = (const float*)d_in[10], *Wff2 = (const float*)d_in[11];
  const float* bq = (const float*)d_in[12], *bk = (const float*)d_in[13];
  const float* bv = (const float*)d_in[14], *bo = (const float*)d_in[15];
  const float* bff1 = (const float*)d_in[16], *bff2 = (const float*)d_in[17];
  const float* ln1b = (const float*)d_in[18], *ln2b = (const float*)d_in[19];
  const float* ln1w = (const float*)d_in[20], *ln2w = (const float*)d_in[21];
  const float* q_W = (const float*)d_in[22];
  const float* ug_W1 = (const float*)d_in[23], *ug_W2 = (const float*)d_in[24];
  const float* k_W = (const float*)d_in[25], *v_W = (const float*)d_in[26];
  const float* fuse_W = (const float*)d_in[27];
  const float* thr_W1 = (const float*)d_in[28], *thr_W2 = (const float*)d_in[29];
  const float* bud_W1 = (const float*)d_in[30], *bud_W2 = (const float*)d_in[31];
  const float* imp_W1 = (const float*)d_in[32], *imp_W2 = (const float*)d_in[33];
  const float* obs_W1 = (const float*)d_in[34], *obs_W2 = (const float*)d_in[35];
  const float* rt_W1 = (const float*)d_in[36], *rt_W2 = (const float*)d_in[37];
  const float* ut_W1 = (const float*)d_in[38], *ut_W2 = (const float*)d_in[39];
  const float* q_b = (const float*)d_in[40];
  const float* ug_b1 = (const float*)d_in[41], *ug_b2 = (const float*)d_in[42];
  const float* k_b = (const float*)d_in[43], *v_b = (const float*)d_in[44];
  const float* fuse_b = (const float*)d_in[45];
  const float* thr_b1 = (const float*)d_in[46], *thr_b2 = (const float*)d_in[47];
  const float* bud_b1 = (const float*)d_in[48], *bud_b2 = (const float*)d_in[49];
  const float* imp_b1 = (const float*)d_in[50], *imp_b2 = (const float*)d_in[51];
  const float* obs_b1 = (const float*)d_in[52], *obs_b2 = (const float*)d_in[53];
  const float* rt_b1 = (const float*)d_in[54], *rt_b2 = (const float*)d_in[55];
  const float* ut_b1 = (const float*)d_in[56], *ut_b2 = (const float*)d_in[57];
  /* schema_keys (d_in[58]) unused by forward */
  const float* schema_vals = (const float*)d_in[59];

  // -------- output layout (flat concatenation, fp32) --------
  float* outp = (float*)d_out;
  const size_t n_logits = (size_t)BL * VV;
  float* out_logits = outp;
  float* out_updk   = outp + n_logits;
  float* out_updv   = out_updk + (size_t)BB * MEMN * DD;
  float* out_age    = out_updv + (size_t)BB * MEMN * DD;
  float* out_scal   = out_age + (size_t)BB * MEMN;  // ereg, l2, util
  float* out_budget = out_scal + 3;

  // -------- workspace carve-out --------
  char* wsb = (char*)d_ws;
  size_t off = 0;
  auto alloc = [&](size_t bytes) -> void* {
    void* p = wsb + off;
    off = (off + bytes + 255) & ~(size_t)255;
    return p;
  };
  const size_t dd2 = (size_t)DD * DD;
  f16* wqT   = (f16*)alloc(NLAYER * dd2 * 2);
  f16* wkT   = (f16*)alloc(NLAYER * dd2 * 2);
  f16* wvT   = (f16*)alloc(NLAYER * dd2 * 2);
  f16* woT   = (f16*)alloc(NLAYER * dd2 * 2);
  f16* wff1T = (f16*)alloc((size_t)NLAYER * DD * 4 * DD * 2);
  f16* wff2T = (f16*)alloc((size_t)NLAYER * DD * 4 * DD * 2);
  f16* embH  = (f16*)alloc((size_t)VV * DD * 2);
  f16* qwT   = (f16*)alloc(dd2 * 2);
  f16* kwT   = (f16*)alloc(dd2 * 2);
  f16* vwT   = (f16*)alloc(dd2 * 2);
  f16* fwT   = (f16*)alloc(dd2 * 2);
  f16* ugw1T = (f16*)alloc((size_t)DD * 64 * 2);
  f16* ugw2T = (f16*)alloc((size_t)DD * 64 * 2);

  const size_t actN = (size_t)BL * DD;
  float* xe_f   = (float*)alloc(actN * 4);
  f16*   xe_h   = (f16*)  alloc(actN * 2);
  float* h_f    = (float*)alloc(actN * 4);
  f16*   h_h    = (f16*)  alloc(actN * 2);
  float* qf     = (float*)alloc(actN * 4);
  float* kf     = (float*)alloc(actN * 4);
  float* vf     = (float*)alloc(actN * 4);
  float* att_f  = (float*)alloc(actN * 4);
  f16*   att_h  = (f16*)  alloc(actN * 2);
  float* tmp_f  = (float*)alloc(actN * 4);
  f16*   ff1_h  = (f16*)  alloc((size_t)BL * 4 * DD * 2);
  f16*   ughid  = (f16*)  alloc((size_t)BL * 64 * 2);
  float* unc_f  = (float*)alloc(actN * 4);
  float* qry_f  = (float*)alloc(actN * 4);
  float* candk  = (float*)alloc(actN * 4);
  float* candv  = (float*)alloc(actN * 4);
  float* ret_f  = (float*)alloc(actN * 4);
  f16*   ret_h  = (f16*)  alloc(actN * 2);
  f16*   fusedh = (f16*)  alloc(actN * 2);
  float* ent    = (float*)alloc((size_t)BL * 4);
  float* ent_n  = (float*)alloc((size_t)BL * 4);
  float* thr    = (float*)alloc((size_t)BL * 4);
  float* stats  = (float*)alloc(BB * 2 * 4);
  float* meanE  = (float*)alloc((size_t)BB * DD * 4);
  float* impraw = (float*)alloc((size_t)BB * NCOMB * 4);
  float* impA   = (float*)alloc((size_t)BB * NCOMB * 4);
  float* impM   = (float*)alloc((size_t)BB * NCOMB * 4);
  float* validf = (float*)alloc((size_t)BB * NCOMB * 4);
  float* obsb   = (float*)alloc((size_t)BB * MEMN * 4);
  float* stats2 = (float*)alloc(BB * 4 * 4);
  int*   topidx = (int*)  alloc((size_t)BB * MEMN * 4);
  float* readk  = (float*)alloc((size_t)BB * NREAD * DD * 4);
  float* readv  = (float*)alloc((size_t)BB * NREAD * DD * 4);
  float* routing= (float*)alloc((size_t)BB * MEMN * SS * 4);
  float* utilb  = (float*)alloc((size_t)BB * SS * 4);

  auto gsz = [](size_t n, int t) -> unsigned { return (unsigned)((n + t - 1) / t); };

  // -------- weight conversions (every call; no cross-call caching) --------
  k_f2h_T<<<gsz(NLAYER * dd2, 256), 256, 0, stream>>>(Wq, wqT, DD, DD, NLAYER);
  k_f2h_T<<<gsz(NLAYER * dd2, 256), 256, 0, stream>>>(Wk, wkT, DD, DD, NLAYER);
  k_f2h_T<<<gsz(NLAYER * dd2, 256), 256, 0, stream>>>(Wv, wvT, DD, DD, NLAYER);
  k_f2h_T<<<gsz(NLAYER * dd2, 256), 256, 0, stream>>>(Wo, woT, DD, DD, NLAYER);
  k_f2h_T<<<gsz((size_t)NLAYER * DD * 4 * DD, 256), 256, 0, stream>>>(Wff1, wff1T, DD, 4 * DD, NLAYER);
  k_f2h_T<<<gsz((size_t)NLAYER * DD * 4 * DD, 256), 256, 0, stream>>>(Wff2, wff2T, 4 * DD, DD, NLAYER);
  k_f2h  <<<gsz((size_t)VV * DD, 256), 256, 0, stream>>>(embed, embH, (size_t)VV * DD);
  k_f2h_T<<<gsz(dd2, 256), 256, 0, stream>>>(q_W, qwT, DD, DD, 1);
  k_f2h_T<<<gsz(dd2, 256), 256, 0, stream>>>(k_W, kwT, DD, DD, 1);
  k_f2h_T<<<gsz(dd2, 256), 256, 0, stream>>>(v_W, vwT, DD, DD, 1);
  k_f2h_T<<<gsz(dd2, 256), 256, 0, stream>>>(fuse_W, fwT, DD, DD, 1);
  k_f2h_T<<<gsz((size_t)DD * 64, 256), 256, 0, stream>>>(ug_W1, ugw1T, DD, 64, 1);
  k_f2h_T<<<gsz((size_t)DD * 64, 256), 256, 0, stream>>>(ug_W2, ugw2T, 64, DD, 1);

  // -------- embedding --------
  k_embed<<<gsz(actN, 256), 256, 0, stream>>>(x, embed, pos, xe_f, xe_h, h_f, h_h);

  // -------- transformer encoder (4 post-norm layers) --------
  for (int i = 0; i < NLAYER; ++i) {
    const f16* wq = wqT + (size_t)i * dd2;
    const f16* wk = wkT + (size_t)i * dd2;
    const f16* wv = wvT + (size_t)i * dd2;
    const f16* wo = woT + (size_t)i * dd2;
    k_gemm<<<gemm_grid(BL, DD), 256, 0, stream>>>(h_h, wq, bq + i * DD, qf, nullptr, BL, DD, DD, 0);
    k_gemm<<<gemm_grid(BL, DD), 256, 0, stream>>>(h_h, wk, bk + i * DD, kf, nullptr, BL, DD, DD, 0);
    k_gemm<<<gemm_grid(BL, DD), 256, 0, stream>>>(h_h, wv, bv + i * DD, vf, nullptr, BL, DD, DD, 0);
    k_attn<<<BB * HH * LL, 128, 0, stream>>>(qf, kf, vf, att_f, att_h);
    k_gemm<<<gemm_grid(BL, DD), 256, 0, stream>>>(att_h, wo, bo + i * DD, tmp_f, nullptr, BL, DD, DD, 0);
    k_add_ln<<<BL, 256, 0, stream>>>(h_f, tmp_f, ln1w + i * DD, ln1b + i * DD, h_f, h_h);
    k_gemm<<<gemm_grid(BL, 4 * DD), 256, 0, stream>>>(
        h_h, wff1T + (size_t)i * DD * 4 * DD, bff1 + i * 4 * DD, nullptr, ff1_h, BL, 4 * DD, DD, 1);
    k_gemm<<<gemm_grid(BL, DD), 256, 0, stream>>>(
        ff1_h, wff2T + (size_t)i * DD * 4 * DD, bff2 + i * DD, tmp_f, nullptr, BL, DD, 4 * DD, 0);
    k_add_ln<<<BL, 256, 0, stream>>>(h_f, tmp_f, ln2w + i * DD, ln2b + i * DD, h_f, h_h);
  }
  // h_f / h_h == encoded

  // -------- base logits (into d_out logits region, used as scratch) --------
  k_gemm<<<gemm_grid(BL, VV), 256, 0, stream>>>(h_h, embH, nullptr, out_logits, nullptr, BL, VV, DD, 0);
  k_entropy<<<BL, 256, 0, stream>>>(out_logits, ent);
  k_entn<<<BB, 256, 0, stream>>>(ent, ent_n, stats);

  // -------- uncertainty gate + queries --------
  k_gemm<<<gemm_grid(BL, 64), 256, 0, stream>>>(h_h, ugw1T, ug_b1, nullptr, ughid, BL, 64, DD, 1);
  k_gemm<<<gemm_grid(BL, DD), 256, 0, stream>>>(ughid, ugw2T, ug_b2, unc_f, nullptr, BL, DD, 64, 2);
  k_gemm<<<gemm_grid(BL, DD), 256, 0, stream>>>(h_h, qwT, q_b, tmp_f, nullptr, BL, DD, DD, 0);
  k_queries<<<gsz(actN, 256), 256, 0, stream>>>(tmp_f, unc_f, qry_f);

  // -------- dynamic threshold + budget --------
  k_thr<<<BL, 64, 0, stream>>>(h_f, stats, thr_W1, thr_b1, thr_W2, thr_b2, thr);
  k_meanL<<<gsz((size_t)BB * DD, 256), 256, 0, stream>>>(h_f, meanE);
  k_mlp512_32_1<<<BB, 32, 0, stream>>>(meanE, bud_W1, bud_b1, bud_W2, bud_b2,
                                       out_budget, BB, 0, 0, 1);

  // -------- candidates + importance --------
  k_gemm<<<gemm_grid(BL, DD), 256, 0, stream>>>(h_h, kwT, k_b, candk, nullptr, BL, DD, DD, 0);
  k_gemm<<<gemm_grid(BL, DD), 256, 0, stream>>>(xe_h, vwT, v_b, candv, nullptr, BL, DD, DD, 0);
  k_mlp512_64_1<<<BB * MEMN, 64, 0, stream>>>(mem_k, obs_W1, obs_b1, obs_W2, obs_b2,
                                              obsb, MEMN, MEMN, 0);
  k_mlp512_64_1<<<BB * MEMN, 64, 0, stream>>>(mem_k, imp_W1, imp_b1, imp_W2, imp_b2,
                                              impraw, MEMN, NCOMB, 0);
  k_mlp512_64_1<<<BB * LL, 64, 0, stream>>>(candk, imp_W1, imp_b1, imp_W2, imp_b2,
                                            impraw, LL, NCOMB, MEMN);
  k_impm<<<gsz((size_t)BB * NCOMB, 256), 256, 0, stream>>>(
      impraw, obsb, mem_age, ent_n, thr, impA, impM, validf);
  k_regstats<<<BB, 256, 0, stream>>>(impM, impA, validf, stats2);

  // -------- top-k select + gather into outputs / read buffers --------
  k_topk<<<BB, 256, 0, stream>>>(impM, topidx);
  k_gather<<<gsz((size_t)BB * MEMN * DD, 256), 256, 0, stream>>>(
      topidx, mem_k, mem_v, mem_age, candk, candv,
      out_updk, out_updv, out_age, readk, readv);

  // -------- schema routing + utility --------
  k_routing<<<BB * MEMN, 64, 0, stream>>>(out_updk, rt_W1, rt_b1, rt_W2, rt_b2, routing);
  k_schema<<<gsz((size_t)BB * SS * DD, 256), 256, 0, stream>>>(
      routing, out_updk, schema_vals, readk, readv);
  k_mlp512_32_1<<<BB * SS, 32, 0, stream>>>(readk, ut_W1, ut_b1, ut_W2, ut_b2,
                                            utilb, SS, NREAD, MEMN, 0);

  // -------- retrieval + fuse + final logits --------
  k_retrieve<<<BL, 128, 0, stream>>>(qry_f, readk, readv, ret_f, ret_h);
  k_gemm<<<gemm_grid(BL, DD), 256, 0, stream>>>(ret_h, fwT, fuse_b, tmp_f, nullptr, BL, DD, DD, 0);
  k_fused<<<gsz(actN, 256), 256, 0, stream>>>(h_f, tmp_f, fusedh);
  k_gemm<<<gemm_grid(BL, VV), 256, 0, stream>>>(fusedh, embH, nullptr, out_logits, nullptr, BL, VV, DD, 0);

  // -------- scalar outputs --------
  k_finalize<<<1, 1, 0, stream>>>(stats2, utilb, out_scal);
}
